// ContinueDecoder_3453153706548
// MI455X (gfx1250) — matrix operand order repaired and checked by reading, not compiled, not run on hardware
//
#include <hip/hip_runtime.h>
#include <hip/hip_bf16.h>
#include <math.h>

// ---------------------------------------------------------------------------
// MI455X (gfx1250) transformer block: bf16 WMMA GEMMs + fused flash attention.
// Wave32, V_WMMA_F32_16X16X32_BF16 everywhere a matmul appears.
// GEMM uses 4x4 register blocking: 64x64 output per wave, 16 WMMA / 32-K step.
// ---------------------------------------------------------------------------

typedef __attribute__((ext_vector_type(16))) __bf16 v16bf;
typedef __attribute__((ext_vector_type(8)))  __bf16 bf16x8;
typedef __attribute__((ext_vector_type(8)))  float  v8f;

constexpr int Bc = 8, Sc = 1024, Dc = 1024, Hc = 16, DFFc = 4096, Cct = 16, Ec = 8;
constexpr int HDc = Dc / Hc;   // 64

#define DEVINL __device__ __forceinline__

DEVINL v16bf cat16(bf16x8 lo, bf16x8 hi) {
  v16bf r;
#pragma unroll
  for (int i = 0; i < 8; ++i) { r[i] = lo[i]; r[i + 8] = hi[i]; }
  return r;
}

// A fragment (16x32, M x K): lane m = lane&15; elements: k = e + 16*hf
// rowp points at the start of this row's 32-element K chunk; koff = 8*(lane>>4).
DEVINL v16bf loadA(const __bf16* rowp, int koff) {
  return cat16(*(const bf16x8*)(rowp + 2 * koff), *(const bf16x8*)(rowp + 2 * koff + 8));
}
// B fragment (32x16, K x N) fed as Bt[n][k]: lane n = lane&15; k = e + 16*hf.
// boff = 16*(lane>>4).
DEVINL v16bf loadB(const __bf16* rowp, int boff) {
  return cat16(*(const bf16x8*)(rowp + boff), *(const bf16x8*)(rowp + boff + 8));
}

DEVINL float gelu_tanh(float x) {
  const float k0 = 0.7978845608028654f, k1 = 0.044715f;
  return 0.5f * x * (1.f + tanhf(k0 * (x + k1 * x * x * x)));
}

// ---------------------------------------------------------------------------
// Generic bf16 WMMA GEMM:  C[m,n] = act(A[m,:]·Bt[n,:] + bias[n]) (+= if accumulate)
// A: [M x K] bf16 (lda), Bt: [N x K] bf16 (ldb), C fp32 / Cbf bf16 (ldc).
// Batched via blockIdx.z; optional eidx[] gates/indexes expert weights.
// Block = 256 threads = 8 waves laid out 2(M) x 4(N); wave tile = 64x64
// (4x4 of 16x16 WMMA tiles, 16 v8f accumulators = 128 VGPRs).
// Macro tile per block: 128 x 256. All dims used are exact multiples.
// ---------------------------------------------------------------------------
__global__ __launch_bounds__(256)
void gemm_bf16_wmma(const __bf16* __restrict__ A, const __bf16* __restrict__ Bt,
                    const float* __restrict__ bias, float* __restrict__ C,
                    __bf16* __restrict__ Cbf,
                    int M, int N, int K, int lda, int ldb, int ldc,
                    int act_gelu, int accumulate,
                    size_t strideA, size_t strideBt, size_t strideBias, size_t strideC,
                    const int* __restrict__ eidx) {
  int bz = blockIdx.z;
  int e = bz;
  if (eidx) { e = eidx[bz]; if (e < 0) return; }
  const __bf16* Ab  = A + (size_t)bz * strideA;
  const __bf16* Btb = Bt + (size_t)e  * strideBt;
  const float*  bb  = bias ? bias + (size_t)e * strideBias : nullptr;

  int wave = threadIdx.x >> 5;
  int lane = threadIdx.x & 31;
  int l16  = lane & 15;
  int hf   = lane >> 4;
  int koff = hf * 8;     // A-fragment K offset
  int boff = hf * 16;    // B-fragment K offset

  int m0 = blockIdx.y * 128 + (wave >> 2) * 64;
  int n0 = blockIdx.x * 256 + (wave & 3) * 64;
  if (m0 >= M || n0 >= N) return;

  const __bf16* ap[4];
  const __bf16* bp[4];
#pragma unroll
  for (int i = 0; i < 4; ++i) {
    ap[i] = Ab  + (size_t)(m0 + i * 16 + l16) * lda;
    bp[i] = Btb + (size_t)(n0 + i * 16 + l16) * ldb;
  }

  v8f acc[4][4] = {};
  for (int k0 = 0; k0 < K; k0 += 32) {
    v16bf av[4], bv[4];
#pragma unroll
    for (int i = 0; i < 4; ++i) av[i] = loadA(ap[i] + k0, koff);
#pragma unroll
    for (int j = 0; j < 4; ++j) bv[j] = loadB(bp[j] + k0, boff);
#pragma unroll
    for (int i = 0; i < 4; ++i)
#pragma unroll
      for (int j = 0; j < 4; ++j)
        acc[i][j] = __builtin_amdgcn_wmma_f32_16x16x32_bf16(
            false, av[i], false, bv[j], (short)0, acc[i][j], false, false);
  }

  float*  Cb   = C   ? C   + (size_t)bz * strideC : nullptr;
  __bf16* Cbfb = Cbf ? Cbf + (size_t)bz * strideC : nullptr;
#pragma unroll
  for (int j = 0; j < 4; ++j) {
    int n = n0 + j * 16 + l16;
    float bval = bb ? bb[n] : 0.f;
#pragma unroll
    for (int i = 0; i < 4; ++i) {
      int mbase = m0 + i * 16 + hf * 8;
#pragma unroll
      for (int r = 0; r < 8; ++r) {
        float v = acc[i][j][r] + bval;
        if (act_gelu) v = gelu_tanh(v);
        size_t idx = (size_t)(mbase + r) * ldc + n;
        if (Cb)   { if (accumulate) v += Cb[idx]; Cb[idx] = v; }
        if (Cbfb) Cbfb[idx] = (__bf16)v;
      }
    }
  }
}

// ---------------------------------------------------------------------------
// Fused flash attention: one wave per (b, h, 16-query tile).
// qkv: [B*S, 3D] bf16;  vt: [B,H,HD,S] bf16 (V pre-transposed);  out: [B*S, D] bf16.
// ---------------------------------------------------------------------------
__global__ __launch_bounds__(256)
void attn_flash_wmma(const __bf16* __restrict__ qkv, const __bf16* __restrict__ vt,
                     __bf16* __restrict__ attn_out) {
  __shared__ __bf16 plds[8 * 16 * 32];   // 1KB per wave: P tile staging (D->A layout)
  int wave = threadIdx.x >> 5;
  int lane = threadIdx.x & 31;
  int l16  = lane & 15;
  int hf   = lane >> 4;
  int koff = hf * 8;
  int boff = hf * 16;

  int qt   = blockIdx.x * 8 + wave;        // 0..8191
  int qrow0 = (qt & 63) * 16;
  int bh    = qt >> 6;
  int hh    = bh & (Hc - 1);
  int b     = bh >> 4;

  const __bf16* qbase = qkv + ((size_t)b * Sc) * (3 * Dc) + hh * HDc;
  const __bf16* kbase = qbase + Dc;
  const __bf16* vtb   = vt + (size_t)bh * HDc * Sc;

  // Q fragments: 16x64 = two 16x32 A fragments
  const __bf16* qp = qbase + (size_t)(qrow0 + l16) * (3 * Dc);
  v16bf aq0 = loadA(qp, koff);
  v16bf aq1 = loadA(qp + 32, koff);

  v8f o0 = {}, o1 = {}, o2 = {}, o3 = {};
  float Mrun[8], Lrun[8];
#pragma unroll
  for (int r = 0; r < 8; ++r) { Mrun[r] = -3.0e38f; Lrun[r] = 0.f; }

  __bf16* pl = plds + wave * (16 * 32);
  const float scale = 0.125f;              // 1/sqrt(64)

  for (int key0 = 0; key0 < Sc; key0 += 32) {
    // scores for 32 keys: two 16x16 tiles, each = 2 WMMA over HD=64
    v8f s0 = {}, s1 = {};
    {
      const __bf16* kp = kbase + (size_t)(key0 + l16) * (3 * Dc);
      s0 = __builtin_amdgcn_wmma_f32_16x16x32_bf16(false, aq0, false, loadB(kp, boff),
                                                   (short)0, s0, false, false);
      s0 = __builtin_amdgcn_wmma_f32_16x16x32_bf16(false, aq1, false, loadB(kp + 32, boff),
                                                   (short)0, s0, false, false);
    }
    {
      const __bf16* kp = kbase + (size_t)(key0 + 16 + l16) * (3 * Dc);
      s1 = __builtin_amdgcn_wmma_f32_16x16x32_bf16(false, aq0, false, loadB(kp, boff),
                                                   (short)0, s1, false, false);
      s1 = __builtin_amdgcn_wmma_f32_16x16x32_bf16(false, aq1, false, loadB(kp + 32, boff),
                                                   (short)0, s1, false, false);
    }
    // online softmax per row (rows live across 16-lane groups)
#pragma unroll
    for (int r = 0; r < 8; ++r) {
      float a0 = s0[r] * scale, a1 = s1[r] * scale;
      float mx = fmaxf(a0, a1);
      for (int off = 1; off < 16; off <<= 1) mx = fmaxf(mx, __shfl_xor(mx, off, 32));
      float mnew = fmaxf(Mrun[r], mx);
      float corr = __expf(Mrun[r] - mnew);
      float p0 = __expf(a0 - mnew), p1 = __expf(a1 - mnew);
      float ps = p0 + p1;
      for (int off = 1; off < 16; off <<= 1) ps += __shfl_xor(ps, off, 32);
      Lrun[r] = Lrun[r] * corr + ps;
      Mrun[r] = mnew;
      o0[r] *= corr; o1[r] *= corr; o2[r] *= corr; o3[r] *= corr;
      int row = r + 8 * hf;
      pl[row * 32 + l16]      = (__bf16)p0;
      pl[row * 32 + l16 + 16] = (__bf16)p1;
    }
    asm volatile("s_wait_dscnt 0" ::: "memory");
    // P as A fragment (16x32), V slices as B fragments (32 keys x 16 hd cols)
    v16bf pa = loadA(pl + l16 * 32, koff);
    o0 = __builtin_amdgcn_wmma_f32_16x16x32_bf16(
        false, pa, false, loadB(vtb + (size_t)(0 * 16 + l16) * Sc + key0, boff),
        (short)0, o0, false, false);
    o1 = __builtin_amdgcn_wmma_f32_16x16x32_bf16(
        false, pa, false, loadB(vtb + (size_t)(1 * 16 + l16) * Sc + key0, boff),
        (short)0, o1, false, false);
    o2 = __builtin_amdgcn_wmma_f32_16x16x32_bf16(
        false, pa, false, loadB(vtb + (size_t)(2 * 16 + l16) * Sc + key0, boff),
        (short)0, o2, false, false);
    o3 = __builtin_amdgcn_wmma_f32_16x16x32_bf16(
        false, pa, false, loadB(vtb + (size_t)(3 * 16 + l16) * Sc + key0, boff),
        (short)0, o3, false, false);
  }

#pragma unroll
  for (int r = 0; r < 8; ++r) {
    float inv = 1.f / Lrun[r];
    int srow = qrow0 + r + 8 * hf;
    __bf16* op = attn_out + ((size_t)b * Sc + srow) * Dc + hh * HDc;
    op[0 * 16 + l16] = (__bf16)(o0[r] * inv);
    op[1 * 16 + l16] = (__bf16)(o1[r] * inv);
    op[2 * 16 + l16] = (__bf16)(o2[r] * inv);
    op[3 * 16 + l16] = (__bf16)(o3[r] * inv);
  }
}

// ---------------------------------------------------------------------------
// Residual + LayerNorm over D=1024: one 256-thread block per row.
// ---------------------------------------------------------------------------
__global__ __launch_bounds__(256)
void resid_ln(const float* __restrict__ a, const float* __restrict__ bres,
              const float* __restrict__ g, const float* __restrict__ be,
              float* __restrict__ of32, __bf16* __restrict__ obf) {
  int row = blockIdx.x;
  const float* ap = a + (size_t)row * Dc;
  const float* bp = bres ? bres + (size_t)row * Dc : nullptr;
  float v[4]; float s = 0.f, ss = 0.f;
#pragma unroll
  for (int i = 0; i < 4; ++i) {
    int d = threadIdx.x + i * 256;
    float x = ap[d] + (bp ? bp[d] : 0.f);
    v[i] = x; s += x; ss += x * x;
  }
  for (int off = 16; off; off >>= 1) { s += __shfl_xor(s, off, 32); ss += __shfl_xor(ss, off, 32); }
  __shared__ float shs[8], shss[8];
  int wave = threadIdx.x >> 5;
  if ((threadIdx.x & 31) == 0) { shs[wave] = s; shss[wave] = ss; }
  __syncthreads();
  s = 0.f; ss = 0.f;
#pragma unroll
  for (int w = 0; w < 8; ++w) { s += shs[w]; ss += shss[w]; }
  float mu = s * (1.f / Dc);
  float var = ss * (1.f / Dc) - mu * mu;
  float rs = rsqrtf(var + 1e-5f);
#pragma unroll
  for (int i = 0; i < 4; ++i) {
    int d = threadIdx.x + i * 256;
    float y = (v[i] - mu) * rs * g[d] + be[d];
    if (of32) of32[(size_t)row * Dc + d] = y;
    if (obf)  obf[(size_t)row * Dc + d] = (__bf16)y;
  }
}

// ---------------------------------------------------------------------------
// Routing: pool over S, distance to centers, argmin -> label + expert idx.
// One block per sample b.
// ---------------------------------------------------------------------------
__global__ __launch_bounds__(256)
void route_kernel(const float* __restrict__ h, const float* __restrict__ centers,
                  const float* __restrict__ radius, const int* __restrict__ c2e,
                  int* __restrict__ eidx_out, float* __restrict__ labels_out) {
  __shared__ float pooled[Dc];
  __shared__ float dist2[Cct];
  int b = blockIdx.x;
  const float* hb = h + (size_t)b * Sc * Dc;
  for (int d = threadIdx.x; d < Dc; d += 256) {
    float s = 0.f;
    for (int s0 = 0; s0 < Sc; ++s0) s += hb[(size_t)s0 * Dc + d];
    pooled[d] = s * (1.f / Sc);
  }
  if (threadIdx.x < Cct) dist2[threadIdx.x] = 0.f;
  __syncthreads();
  for (int c = 0; c < Cct; ++c) {
    float acc = 0.f;
    for (int d = threadIdx.x; d < Dc; d += 256) {
      float df = pooled[d] - centers[(size_t)c * Dc + d];
      acc += df * df;
    }
    for (int off = 16; off; off >>= 1) acc += __shfl_xor(acc, off, 32);
    if ((threadIdx.x & 31) == 0) atomicAdd(&dist2[c], acc);
  }
  __syncthreads();
  if (threadIdx.x == 0) {
    int best = 0; float bd = 3.0e38f;
    for (int c = 0; c < Cct; ++c) {
      float dd = sqrtf(dist2[c]) / radius[c];
      if (dd < bd) { bd = dd; best = c; }
    }
    eidx_out[b] = c2e[best];
    labels_out[b] = (float)best;
  }
}

// ---------------------------------------------------------------------------
// Data movement helpers
// ---------------------------------------------------------------------------
__global__ void f32_to_bf16(const float* __restrict__ in, __bf16* __restrict__ out, size_t n) {
  size_t i = (size_t)blockIdx.x * 256 + threadIdx.x;
  if (i < n) out[i] = (__bf16)in[i];
}

// in: [R x Cc] fp32 -> out: [Cc x R] bf16 (batched)
__global__ __launch_bounds__(256)
void transpose_to_bf16(const float* __restrict__ in, __bf16* __restrict__ out,
                       int R, int Ccol, size_t inS, size_t outS) {
  __shared__ float tile[32][33];
  const float* inb = in + (size_t)blockIdx.z * inS;
  __bf16* outb = out + (size_t)blockIdx.z * outS;
  int c0 = blockIdx.x * 32, r0 = blockIdx.y * 32;
  int tx = threadIdx.x & 31, ty = threadIdx.x >> 5;
#pragma unroll
  for (int i = 0; i < 32; i += 8) {
    int r = r0 + ty + i, c = c0 + tx;
    if (r < R && c < Ccol) tile[ty + i][tx] = inb[(size_t)r * Ccol + c];
  }
  __syncthreads();
#pragma unroll
  for (int i = 0; i < 32; i += 8) {
    int c = c0 + ty + i, r = r0 + tx;
    if (c < Ccol && r < R) outb[(size_t)c * R + r] = (__bf16)tile[tx][ty + i];
  }
}

// vt[(b*H+h)*HD*S + col*S + s] = qkv[(b*S+s)*3D + 2D + h*HD + col]
__global__ void build_vt(const __bf16* __restrict__ qkv, __bf16* __restrict__ vt) {
  size_t i = (size_t)blockIdx.x * 256 + threadIdx.x;  // over B*H*HD*S
  int s = (int)(i & (size_t)(Sc - 1));
  size_t t = i >> 10;
  int col = (int)(t & (HDc - 1)); t >>= 6;
  int h = (int)(t & (Hc - 1));
  int b = (int)(t >> 4);
  vt[i] = qkv[((size_t)b * Sc + s) * (3 * Dc) + 2 * Dc + (size_t)h * HDc + col];
}

// ---------------------------------------------------------------------------
extern "C" void kernel_launch(void* const* d_in, const int* in_sizes, int n_in,
                              void* d_out, int out_size, void* d_ws, size_t ws_size,
                              hipStream_t stream) {
  (void)in_sizes; (void)n_in; (void)out_size; (void)ws_size;
  const float* x      = (const float*)d_in[0];
  const float* Wqkv   = (const float*)d_in[1];
  const float* bqkv   = (const float*)d_in[2];
  const float* Wo     = (const float*)d_in[3];
  const float* bo     = (const float*)d_in[4];
  const float* ln1_g  = (const float*)d_in[5];
  const float* ln1_b  = (const float*)d_in[6];
  const float* W1     = (const float*)d_in[7];
  const float* b1     = (const float*)d_in[8];
  const float* W2     = (const float*)d_in[9];
  const float* b2     = (const float*)d_in[10];
  const float* eW1    = (const float*)d_in[11];
  const float* eb1    = (const float*)d_in[12];
  const float* eW2    = (const float*)d_in[13];
  const float* eb2    = (const float*)d_in[14];
  const float* ln2_g  = (const float*)d_in[15];
  const float* ln2_b  = (const float*)d_in[16];
  const float* centers= (const float*)d_in[17];
  const float* radius = (const float*)d_in[18];
  const int*   c2e    = (const int*)d_in[19];
  float* out = (float*)d_out;

  char* w = (char*)d_ws;
  size_t off = 0;
  auto alloc = [&](size_t bytes) -> void* {
    void* p = w + off;
    off = (off + bytes + 255) & ~(size_t)255;
    return p;
  };
  const int MS = Bc * Sc;  // 8192
  __bf16* x_bf    = (__bf16*)alloc((size_t)MS * Dc * 2);
  __bf16* Wqkvt   = (__bf16*)alloc((size_t)3 * Dc * Dc * 2);
  __bf16* Wot     = (__bf16*)alloc((size_t)Dc * Dc * 2);
  __bf16* W1t     = (__bf16*)alloc((size_t)DFFc * Dc * 2);
  __bf16* W2t     = (__bf16*)alloc((size_t)Dc * DFFc * 2);
  __bf16* eW1t    = (__bf16*)alloc((size_t)Ec * DFFc * Dc * 2);
  __bf16* eW2t    = (__bf16*)alloc((size_t)Ec * Dc * DFFc * 2);
  __bf16* qkv_bf  = (__bf16*)alloc((size_t)MS * 3 * Dc * 2);
  __bf16* vt      = (__bf16*)alloc((size_t)Bc * Hc * HDc * Sc * 2);
  __bf16* attn_bf = (__bf16*)alloc((size_t)MS * Dc * 2);
  float*  aproj   = (float*) alloc((size_t)MS * Dc * 4);
  float*  h_f32   = (float*) alloc((size_t)MS * Dc * 4);
  __bf16* h_bf    = (__bf16*)alloc((size_t)MS * Dc * 2);
  __bf16* mid_bf  = (__bf16*)alloc((size_t)MS * DFFc * 2);
  float*  ffn     = (float*) alloc((size_t)MS * Dc * 4);
  int*    eidx    = (int*)   alloc(256);

  dim3 blk(256);

  // --- precision / layout prep ---
  f32_to_bf16<<<(unsigned)(((size_t)MS * Dc + 255) / 256), blk, 0, stream>>>(x, x_bf, (size_t)MS * Dc);
  transpose_to_bf16<<<dim3(3 * Dc / 32, Dc / 32, 1), blk, 0, stream>>>(Wqkv, Wqkvt, Dc, 3 * Dc, 0, 0);
  transpose_to_bf16<<<dim3(Dc / 32, Dc / 32, 1), blk, 0, stream>>>(Wo, Wot, Dc, Dc, 0, 0);
  transpose_to_bf16<<<dim3(DFFc / 32, Dc / 32, 1), blk, 0, stream>>>(W1, W1t, Dc, DFFc, 0, 0);
  transpose_to_bf16<<<dim3(Dc / 32, DFFc / 32, 1), blk, 0, stream>>>(W2, W2t, DFFc, Dc, 0, 0);
  transpose_to_bf16<<<dim3(DFFc / 32, Dc / 32, Ec), blk, 0, stream>>>(
      eW1, eW1t, Dc, DFFc, (size_t)Dc * DFFc, (size_t)DFFc * Dc);
  transpose_to_bf16<<<dim3(Dc / 32, DFFc / 32, Ec), blk, 0, stream>>>(
      eW2, eW2t, DFFc, Dc, (size_t)DFFc * Dc, (size_t)Dc * DFFc);

  // --- QKV projection: [8192 x 3072] = x_bf @ Wqkv ---
  gemm_bf16_wmma<<<dim3(3 * Dc / 256, MS / 128, 1), blk, 0, stream>>>(
      x_bf, Wqkvt, bqkv, nullptr, qkv_bf, MS, 3 * Dc, Dc, Dc, Dc, 3 * Dc,
      0, 0, 0, 0, 0, 0, nullptr);

  // --- attention ---
  build_vt<<<(unsigned)(((size_t)Bc * Hc * HDc * Sc) / 256), blk, 0, stream>>>(qkv_bf, vt);
  attn_flash_wmma<<<dim3(Bc * Hc * (Sc / 16) / 8), blk, 0, stream>>>(qkv_bf, vt, attn_bf);

  // --- output projection + residual LN -> h ---
  gemm_bf16_wmma<<<dim3(Dc / 256, MS / 128, 1), blk, 0, stream>>>(
      attn_bf, Wot, bo, aproj, nullptr, MS, Dc, Dc, Dc, Dc, Dc,
      0, 0, 0, 0, 0, 0, nullptr);
  resid_ln<<<MS, blk, 0, stream>>>(x, aproj, ln1_g, ln1_b, h_f32, h_bf);

  // --- routing (labels + expert idx) ---
  route_kernel<<<Bc, blk, 0, stream>>>(h_f32, centers, radius, c2e, eidx,
                                       out + (size_t)MS * Dc);

  // --- base FFN ---
  gemm_bf16_wmma<<<dim3(DFFc / 256, MS / 128, 1), blk, 0, stream>>>(
      h_bf, W1t, b1, nullptr, mid_bf, MS, DFFc, Dc, Dc, Dc, DFFc,
      1, 0, 0, 0, 0, 0, nullptr);
  gemm_bf16_wmma<<<dim3(Dc / 256, MS / 128, 1), blk, 0, stream>>>(
      mid_bf, W2t, b2, ffn, nullptr, MS, Dc, DFFc, DFFc, DFFc, Dc,
      0, 0, 0, 0, 0, 0, nullptr);

  // --- expert FFN (batched over b, gated by eidx, accumulated into ffn) ---
  gemm_bf16_wmma<<<dim3(DFFc / 256, Sc / 128, Bc), blk, 0, stream>>>(
      h_bf, eW1t, eb1, nullptr, mid_bf, Sc, DFFc, Dc, Dc, Dc, DFFc,
      1, 0, (size_t)Sc * Dc, (size_t)DFFc * Dc, (size_t)DFFc, (size_t)Sc * DFFc, eidx);
  gemm_bf16_wmma<<<dim3(Dc / 256, Sc / 128, Bc), blk, 0, stream>>>(
      mid_bf, eW2t, eb2, ffn, nullptr, Sc, Dc, DFFc, DFFc, DFFc, Dc,
      0, 1, (size_t)Sc * DFFc, (size_t)Dc * DFFc, (size_t)Dc, (size_t)Sc * Dc, eidx);

  // --- final residual LN -> out ---
  resid_ln<<<MS, blk, 0, stream>>>(h_f32, ffn, ln2_g, ln2_b, out, nullptr);
}
